// GAT_38852274160025
// MI455X (gfx1250) — matrix operand order repaired and checked by reading, not compiled, not run on hardware
//
#include <hip/hip_runtime.h>

// ---------------------------------------------------------------------------
// 3-layer GAT for MI455X (gfx1250, wave32).
//   GEMMs  : v_wmma_f32_16x16x32_f16, one wave per 16-row stripe computing all
//            output columns (A held in registers, W^T streamed from L2)
//   Edges  : 3-pass segment softmax with L2-resident fp32 atomics
// ---------------------------------------------------------------------------

typedef __attribute__((ext_vector_type(16))) _Float16 v16h;
typedef __attribute__((ext_vector_type(8)))  _Float16 v8h;
typedef __attribute__((ext_vector_type(8)))  float    v8f;

#define NEG_SLOPE 0.2f
#define EPS_F     1e-16f
#define FIN       128      // K dimension of every GEMM in this model

__device__ __forceinline__ void atomicMaxFloat(float* addr, float v) {
    // init value is -inf; standard monotone int/uint punning trick
    if (v >= 0.0f) atomicMax((int*)addr, __float_as_int(v));
    else           atomicMin((unsigned int*)addr, __float_as_uint(v));
}

// ---------------------------- utility kernels ------------------------------

__global__ void fill_f32_k(float* __restrict__ p, float v, int n) {
    int t = blockIdx.x * blockDim.x + threadIdx.x;
    if (t < n) p[t] = v;
}

__global__ void cvt_f16_k(const float* __restrict__ s, _Float16* __restrict__ d, int n) {
    int t = blockIdx.x * blockDim.x + threadIdx.x;
    if (t < n) d[t] = (_Float16)s[t];
}

// W[K,Nc] (row-major f32) -> Wt[Nc,K] f16 so B-operand loads are contiguous in K
__global__ void transpose_f16_k(const float* __restrict__ W, _Float16* __restrict__ Wt,
                                int K, int Nc) {
    int t = blockIdx.x * blockDim.x + threadIdx.x;
    if (t >= K * Nc) return;
    int k = t / Nc, n = t - k * Nc;
    Wt[n * K + k] = (_Float16)W[t];
}

// ------------------------------- WMMA GEMM ---------------------------------
// D[N,Fout] = A16[N,FIN] * W with Bt16[Fout,FIN] = W^T (f16 in, f32 acc).
// One wave32 owns a 16-row stripe: A (16xFIN) is loaded once into 4 v16h
// registers, then the wave sweeps all Fout/16 column tiles with 4 WMMAs each.
// Operand register layouts follow CDNA5 ISA §7.12.2.
__global__ void gemm_wmma_f16_k(const _Float16* __restrict__ A16,
                                const _Float16* __restrict__ Bt16,
                                float* __restrict__ D,
                                int N, int Fout) {
    int gt   = blockIdx.x * blockDim.x + threadIdx.x;
    int wave = gt >> 5;
    int lane = gt & 31;
    int rowTiles = (N + 15) >> 4;
    if (wave >= rowTiles) return;
    int row0 = wave << 4;
    int lm = lane & 15;
    int hi = lane >> 4;                       // half-wave select
    // clamp OOB rows to a valid row (loads stay legal; stores are guarded)
    int arow = row0 + lm;
    if (arow >= N) arow = N - 1;

    // A lanes 0-15 hold K {0..7,16..23}(+32k); lanes 16-31 hold K {8..15,24..31}
    const _Float16* ap = A16 + (size_t)arow * FIN + (hi << 3);
    v16h a[FIN / 32];
    #pragma unroll
    for (int k = 0; k < FIN / 32; ++k) {
        v8h alo = *(const v8h*)(ap + 32 * k);
        v8h ahi = *(const v8h*)(ap + 32 * k + 16);
        #pragma unroll
        for (int i = 0; i < 8; ++i) { a[k][i] = alo[i]; a[k][8 + i] = ahi[i]; }
    }

    // wave-uniform "full stripe" predicate -> scalar branch, no exec masking
    bool fullTile = (__builtin_amdgcn_readfirstlane(row0) + 16 <= N);

    int colTiles = Fout >> 4;
    int mbase = row0 + (hi << 3);
    for (int ct = 0; ct < colTiles; ++ct) {
        // B lanes 0-15 hold K {0..7,16..23}(+32k); lanes 16-31 hold K {8..15,24..31}
        const _Float16* bp = Bt16 + (size_t)((ct << 4) + lm) * FIN + (hi << 3);
        v8f c = {};
        #pragma unroll
        for (int k = 0; k < FIN / 32; ++k) {
            v16h b;
            v8h blo = *(const v8h*)(bp + 32 * k);
            v8h bhi = *(const v8h*)(bp + 32 * k + 16);
            #pragma unroll
            for (int i = 0; i < 8; ++i) { b[i] = blo[i]; b[8 + i] = bhi[i]; }
            c = __builtin_amdgcn_wmma_f32_16x16x32_f16(false, a[k], false, b,
                                                       (short)0, c, false, false);
        }
        // C/D layout: VGPR r -> M = r + 8*hi, N = lane&15
        int ncol = (ct << 4) + lm;
        float* dp = D + (size_t)mbase * Fout + ncol;
        if (fullTile) {
            #pragma unroll
            for (int r = 0; r < 8; ++r) dp[(size_t)r * Fout] = c[r];
        } else {
            #pragma unroll
            for (int r = 0; r < 8; ++r)
                if (mbase + r < N) dp[(size_t)r * Fout] = c[r];
        }
    }
}

// ------------------------- per-node attention dots -------------------------
// As[n,h] = <h[n,h,:], a_src[h,:]> ; Ad likewise
__global__ void alpha_k(const float* __restrict__ Hf,
                        const float* __restrict__ a_s, const float* __restrict__ a_d,
                        float* __restrict__ As, float* __restrict__ Ad,
                        int N, int Hh, int C) {
    int t = blockIdx.x * blockDim.x + threadIdx.x;
    if (t >= N * Hh) return;
    int n = t / Hh, h = t - n * Hh;
    const float* hp = Hf + (size_t)n * Hh * C + h * C;
    float ss = 0.f, sd = 0.f;
    for (int c = 0; c < C; ++c) {
        float v = hp[c];
        ss += v * a_s[h * C + c];
        sd += v * a_d[h * C + c];
    }
    As[t] = ss; Ad[t] = sd;
}

// ------------------------------ edge passes --------------------------------

__global__ void edge_max_k(const int* __restrict__ ei, int E, int N, int Hh,
                           const float* __restrict__ As, const float* __restrict__ Ad,
                           float* __restrict__ Ew, float* __restrict__ Mx) {
    int e = blockIdx.x * blockDim.x + threadIdx.x;
    if (e >= E + N) return;
    int s, d;
    if (e < E) { s = ei[e]; d = ei[E + e]; } else { s = e - E; d = s; }
    for (int h = 0; h < Hh; ++h) {
        float v = As[s * Hh + h] + Ad[d * Hh + h];
        v = (v > 0.f) ? v : NEG_SLOPE * v;            // leaky_relu
        Ew[(size_t)e * Hh + h] = v;
        atomicMaxFloat(&Mx[d * Hh + h], v);
    }
}

__global__ void edge_exp_k(const int* __restrict__ ei, int E, int N, int Hh,
                           const float* __restrict__ Mx,
                           float* __restrict__ Ew, float* __restrict__ Dn) {
    int e = blockIdx.x * blockDim.x + threadIdx.x;
    if (e >= E + N) return;
    int d = (e < E) ? ei[E + e] : (e - E);
    for (int h = 0; h < Hh; ++h) {
        float w = expf(Ew[(size_t)e * Hh + h] - Mx[d * Hh + h]);
        Ew[(size_t)e * Hh + h] = w;
        atomicAdd(&Dn[d * Hh + h], w);
    }
}

// Dn <- 1/(Dn + eps) so the big aggregate pass multiplies instead of divides
__global__ void recip_k(float* __restrict__ Dn, int n) {
    int t = blockIdx.x * blockDim.x + threadIdx.x;
    if (t < n) Dn[t] = 1.0f / (Dn[t] + EPS_F);
}

// one thread per (edge, 4-channel group); a wave covers one edge (Fout=128)
__global__ void edge_agg_k(const int* __restrict__ ei, int E, int N, int Hh,
                           int C, int Fout,
                           const float* __restrict__ Ew, const float* __restrict__ DnR,
                           const float* __restrict__ Hf, float* __restrict__ Acc) {
    int t = blockIdx.x * blockDim.x + threadIdx.x;
    int quads = Fout >> 2;
    int e = t / quads;
    if (e >= E + N) return;
    int q = t - e * quads;
    int s, d;
    if (e < E) { s = ei[e]; d = ei[E + e]; } else { s = e - E; d = s; }
    int c0 = q << 2;
    int h  = c0 / C;
    float alpha = Ew[(size_t)e * Hh + h] * DnR[d * Hh + h];
    const float4 hv = *(const float4*)(Hf + (size_t)s * Fout + c0);
    float* op = Acc + (size_t)d * Fout + c0;
    atomicAdd(op + 0, alpha * hv.x);
    atomicAdd(op + 1, alpha * hv.y);
    atomicAdd(op + 2, alpha * hv.z);
    atomicAdd(op + 3, alpha * hv.w);
}

// ------------------------------- epilogues ---------------------------------

__global__ void bias_relu_f16_k(const float* __restrict__ Acc, const float* __restrict__ b,
                                _Float16* __restrict__ Xh, int total, int F) {
    int t = blockIdx.x * blockDim.x + threadIdx.x;
    if (t >= total) return;
    int c = t % F;
    float v = Acc[t] + b[c];
    Xh[t] = (_Float16)(v > 0.f ? v : 0.f);
}

__global__ void bias_out_k(const float* __restrict__ Acc, const float* __restrict__ b,
                           float* __restrict__ out, int total, int F) {
    int t = blockIdx.x * blockDim.x + threadIdx.x;
    if (t >= total) return;
    int c = t % F;
    out[t] = Acc[t] + b[c];
}

// ------------------------------- launcher ----------------------------------

extern "C" void kernel_launch(void* const* d_in, const int* in_sizes, int n_in,
                              void* d_out, int out_size, void* d_ws, size_t ws_size,
                              hipStream_t stream) {
    (void)n_in; (void)ws_size;
    const float* x   = (const float*)d_in[0];
    const int*   ei  = (const int*)  d_in[1];
    const float* W0  = (const float*)d_in[2];
    const float* as0 = (const float*)d_in[3];
    const float* ad0 = (const float*)d_in[4];
    const float* b0  = (const float*)d_in[5];
    const float* W1  = (const float*)d_in[6];
    const float* as1 = (const float*)d_in[7];
    const float* ad1 = (const float*)d_in[8];
    const float* b1  = (const float*)d_in[9];
    const float* W2  = (const float*)d_in[10];
    const float* as2 = (const float*)d_in[11];
    const float* ad2 = (const float*)d_in[12];
    const float* b2  = (const float*)d_in[13];
    float* out = (float*)d_out;

    const int HID = 64;
    const int N   = out_size / HID;        // 50000
    const int E   = in_sizes[1] / 2;       // 800000
    const int ET  = E + N;                 // + self loops
    const int F0  = in_sizes[0] / N;       // 128 (== FIN)
    const int F   = 128;                   // heads*hid (layers 0/1)

    // ---- workspace carve-out (256B-aligned slices of d_ws) ----
    char* wp = (char*)d_ws;
    auto carve = [&](size_t bytes) {
        void* r = (void*)wp;
        wp += (bytes + 255) & ~(size_t)255;
        return r;
    };
    _Float16* Xh  = (_Float16*)carve((size_t)N * 128 * sizeof(_Float16)); // layer input (f16)
    _Float16* Wt0 = (_Float16*)carve((size_t)F0 * 128 * sizeof(_Float16));
    _Float16* Wt1 = (_Float16*)carve((size_t)128 * 128 * sizeof(_Float16));
    _Float16* Wt2 = (_Float16*)carve((size_t)128 * 64  * sizeof(_Float16));
    float* Hf  = (float*)carve((size_t)N * 128 * sizeof(float));          // h = x@W
    float* Acc = (float*)carve((size_t)N * 128 * sizeof(float));          // segment-sum acc
    float* As  = (float*)carve((size_t)N * 2 * sizeof(float));
    float* Ad  = (float*)carve((size_t)N * 2 * sizeof(float));
    float* Mx  = (float*)carve((size_t)N * 2 * sizeof(float));
    float* Dn  = (float*)carve((size_t)N * 2 * sizeof(float));
    float* Ew  = (float*)carve((size_t)ET * 2 * sizeof(float));           // edge e / exp(e-m)

    const int B = 256;
    auto nb = [&](long long n) { return (int)((n + B - 1) / B); };

    // ---- prologue: precision conversion + weight transpose ----
    cvt_f16_k<<<nb((long long)N * F0), B, 0, stream>>>(x, Xh, N * F0);
    transpose_f16_k<<<nb(F0 * 128), B, 0, stream>>>(W0, Wt0, F0, 128);
    transpose_f16_k<<<nb(128 * 128), B, 0, stream>>>(W1, Wt1, 128, 128);
    transpose_f16_k<<<nb(128 * 64),  B, 0, stream>>>(W2, Wt2, 128, 64);

    auto run_layer = [&](const _Float16* Wt, int Fout, int Hh,
                         const float* a_s, const float* a_d, const float* bias,
                         bool last) {
        int C = Fout / Hh;
        // GEMM: one wave per 16-row stripe (A in regs, 8 waves per block)
        int stripes = (N + 15) / 16;
        gemm_wmma_f16_k<<<(stripes + 7) / 8, B, 0, stream>>>(Xh, Wt, Hf, N, Fout);
        // attention logits per node
        alpha_k<<<nb((long long)N * Hh), B, 0, stream>>>(Hf, a_s, a_d, As, Ad, N, Hh, C);
        // re-init segment buffers every call (determinism)
        fill_f32_k<<<nb((long long)N * Hh), B, 0, stream>>>(Mx, -__builtin_inff(), N * Hh);
        fill_f32_k<<<nb((long long)N * Hh), B, 0, stream>>>(Dn, 0.f, N * Hh);
        fill_f32_k<<<nb((long long)N * Fout), B, 0, stream>>>(Acc, 0.f, N * Fout);
        // segment softmax over edges + scatter aggregate
        edge_max_k<<<nb(ET), B, 0, stream>>>(ei, E, N, Hh, As, Ad, Ew, Mx);
        edge_exp_k<<<nb(ET), B, 0, stream>>>(ei, E, N, Hh, Mx, Ew, Dn);
        recip_k<<<nb((long long)N * Hh), B, 0, stream>>>(Dn, N * Hh);
        long long aggThreads = (long long)ET * (Fout >> 2);
        edge_agg_k<<<nb(aggThreads), B, 0, stream>>>(ei, E, N, Hh, C, Fout, Ew, Dn, Hf, Acc);
        // epilogue
        if (!last)
            bias_relu_f16_k<<<nb((long long)N * Fout), B, 0, stream>>>(Acc, bias, Xh, N * Fout, Fout);
        else
            bias_out_k<<<nb((long long)N * Fout), B, 0, stream>>>(Acc, bias, out, N * Fout, Fout);
    };

    run_layer(Wt0, F,  2, as0, ad0, b0, false);
    run_layer(Wt1, F,  2, as1, ad1, b1, false);
    run_layer(Wt2, HID, 1, as2, ad2, b2, true);
}
